// Rwkv6SelfAttention_36661840839089
// MI455X (gfx1250) — matrix operand order repaired and checked by reading, not compiled, not run on hardware
//
#include <hip/hip_runtime.h>

#define T_LEN 2048
#define H_DIM 2048
#define NHEAD 32
#define HSZ   64
#define DMIXC 32
#define DDECC 64
#define EPSV  1e-5f

typedef __attribute__((ext_vector_type(16))) __bf16 v16bf;
typedef __attribute__((ext_vector_type(8)))  __bf16 v8bf;
typedef __attribute__((ext_vector_type(8)))  float  v8f;

// ---------------------------------------------------------------------------
// Generic bf16 WMMA GEMM:  C[M,N] = act(A[M,K] * B^T)  with B stored N x K.
// Block = 128 threads (4 waves, 2x2), block tile 128x128, wave tile 64x64,
// K-step 32 -> v_wmma_f32_16x16x32_bf16, software-pipelined (double-buffered
// fragments) so the next K-slab's global_load_b128s overlap the current
// 16-WMMA pack. Edge tiles are handled by clamping fragment base pointers
// (loop-invariant, no in-loop EXEC manipulation); stores remain guarded.
// ACT: 0=none, 1=tanh, 2=silu.  ADD_RES adds fp32 residual. OUT_BF16 selects
// output dtype.
// ---------------------------------------------------------------------------
template <int ACT, bool ADD_RES, bool OUT_BF16>
__global__ __launch_bounds__(128) void gemm_wmma(
    const __bf16* __restrict__ A, int lda,
    const __bf16* __restrict__ B, int ldb,   // B is N x K (row-major)
    void* __restrict__ Cv, int ldc,
    const float* __restrict__ res,
    int M, int N, int K)
{
    const int lane = threadIdx.x & 31;
    const int wave = threadIdx.x >> 5;
    const int wm   = wave >> 1;
    const int wn   = wave & 1;
    const int m_base = blockIdx.y * 128 + wm * 64;
    const int n_base = blockIdx.x * 128 + wn * 64;
    const int lrow = lane & 15;   // row (A) / col (B) within 16x16 tile
    const int lhi  = lane >> 4;   // lane-group select per ISA layout

    // Fragment base pointers, clamped so edge tiles read valid (but unused)
    // memory. A layout: lane holds row m, 16 contiguous K at lhi*16.
    // B layout (N x K storage): lane holds col n, 16 contiguous K at lhi*16.
    const __bf16* ap[4];
    const __bf16* bp[4];
#pragma unroll
    for (int i = 0; i < 4; ++i) {
        int m = m_base + i * 16; if (m > M - 16) m = M - 16;
        ap[i] = A + (size_t)(m + lrow) * lda + lhi * 16;
    }
#pragma unroll
    for (int j = 0; j < 4; ++j) {
        int n = n_base + j * 16; if (n > N - 16) n = N - 16;
        bp[j] = B + (size_t)(n + lrow) * ldb + lhi * 16;
    }

    v8f acc[4][4] = {};
    v16bf aX[4], bX[4], aY[4], bY[4];

    auto loadA = [&](v16bf* dst, int k) {
#pragma unroll
        for (int i = 0; i < 4; ++i) {
            v8bf lo = *(const v8bf*)(ap[i] + k);
            v8bf hi = *(const v8bf*)(ap[i] + k + 8);
#pragma unroll
            for (int e = 0; e < 8; ++e) { dst[i][e] = lo[e]; dst[i][8 + e] = hi[e]; }
        }
    };
    auto loadB = [&](v16bf* dst, int k) {
#pragma unroll
        for (int j = 0; j < 4; ++j) dst[j] = *(const v16bf*)(bp[j] + k);
    };
    auto mma = [&](const v16bf* af, const v16bf* bf) {
#pragma unroll
        for (int i = 0; i < 4; ++i)
#pragma unroll
            for (int j = 0; j < 4; ++j)
                acc[i][j] = __builtin_amdgcn_wmma_f32_16x16x32_bf16(
                    false, af[i], false, bf[j], (short)0, acc[i][j], false, false);
    };

    loadA(aX, 0);
    loadB(bX, 0);
    int k0 = 0;
    for (;;) {
        if (k0 + 32 < K) { loadA(aY, k0 + 32); loadB(bY, k0 + 32); }
        if (k0 + 64 < K) __builtin_prefetch((const void*)(ap[0] + k0 + 64), 0, 3);
        mma(aX, bX);
        k0 += 32;
        if (k0 >= K) break;
        if (k0 + 32 < K) { loadA(aX, k0 + 32); loadB(bX, k0 + 32); }
        mma(aY, bY);
        k0 += 32;
        if (k0 >= K) break;
    }

    // Epilogue. C/D layout: VGPR r -> row m_base+i*16 + lhi*8 + r, col lrow.
#pragma unroll
    for (int i = 0; i < 4; ++i) {
        const int mrow0 = m_base + i * 16 + lhi * 8;
#pragma unroll
        for (int j = 0; j < 4; ++j) {
            const int n = n_base + j * 16 + lrow;
            if (n >= N) continue;
#pragma unroll
            for (int rI = 0; rI < 8; ++rI) {
                const int m = mrow0 + rI;
                if (m >= M) continue;
                float v = acc[i][j][rI];
                if (ACT == 1)      v = tanhf(v);
                else if (ACT == 2) v = v / (1.0f + __expf(-v));
                if (ADD_RES) v += res[(size_t)m * ldc + n];
                if (OUT_BF16) ((__bf16*)Cv)[(size_t)m * ldc + n] = (__bf16)v;
                else          ((float*)Cv)[(size_t)m * ldc + n]  = v;
            }
        }
    }
}

// ---------------------------------------------------------------------------
// LayerNorm + token shift. One block per row t. Computes xn, sx = prev - xn,
// and xxx = xn + sx * time_maa_x (bf16 for the first GEMM). Recomputes
// LN(x[t-1]) locally to avoid inter-block deps.
// ---------------------------------------------------------------------------
__global__ __launch_bounds__(256) void ln_shift_kernel(
    const float* __restrict__ x, const float* __restrict__ state1,
    const float* __restrict__ w, const float* __restrict__ b,
    const float* __restrict__ tmx,
    float* __restrict__ xn, float* __restrict__ sx,
    __bf16* __restrict__ xxx_bf, float* __restrict__ state1_out)
{
    const int t = blockIdx.x;
    const int tid = threadIdx.x;
    __shared__ float red[256];

    const float* xc = x + (size_t)t * H_DIM;
    const float* xp = x + (size_t)(t - 1) * H_DIM;

    float vc[8], vp[8];
    float s_c = 0.f, q_c = 0.f, s_p = 0.f, q_p = 0.f;
#pragma unroll
    for (int i = 0; i < 8; ++i) {
        const int e = tid + i * 256;
        float a = xc[e]; vc[i] = a; s_c += a; q_c += a * a;
        if (t > 0) { float p = xp[e]; vp[i] = p; s_p += p; q_p += p * p; }
        else vp[i] = 0.f;
    }

    auto blockReduce = [&](float val) -> float {
        red[tid] = val; __syncthreads();
        for (int s = 128; s > 0; s >>= 1) {
            if (tid < s) red[tid] += red[tid + s];
            __syncthreads();
        }
        float r = red[0]; __syncthreads();
        return r;
    };

    const float sumc = blockReduce(s_c);
    const float sumq = blockReduce(q_c);
    const float mu   = sumc / H_DIM;
    const float rstd = rsqrtf(sumq / H_DIM - mu * mu + EPSV);

    float mup = 0.f, rstdp = 0.f;
    if (t > 0) {
        const float sp = blockReduce(s_p);
        const float qp = blockReduce(q_p);
        mup   = sp / H_DIM;
        rstdp = rsqrtf(qp / H_DIM - mup * mup + EPSV);
    }

#pragma unroll
    for (int i = 0; i < 8; ++i) {
        const int e = tid + i * 256;
        const float xnv  = (vc[i] - mu) * rstd * w[e] + b[e];
        const float prev = (t > 0) ? ((vp[i] - mup) * rstdp * w[e] + b[e]) : state1[e];
        const float sxv  = prev - xnv;
        const size_t idx = (size_t)t * H_DIM + e;
        xn[idx] = xnv;
        sx[idx] = sxv;
        xxx_bf[idx] = (__bf16)(xnv + sxv * tmx[e]);
        if (t == T_LEN - 1) state1_out[e] = xnv;
    }
}

// mixed = sx * (tmp + bias) + xn  -> bf16
__global__ void mixed_kernel(const float* __restrict__ tmp,
                             const float* __restrict__ bias,
                             const float* __restrict__ sx,
                             const float* __restrict__ xn,
                             __bf16* __restrict__ outbf)
{
    const size_t idx = (size_t)blockIdx.x * blockDim.x + threadIdx.x;
    if (idx >= (size_t)T_LEN * H_DIM) return;
    const int e = (int)(idx & (H_DIM - 1));
    outbf[idx] = (__bf16)(sx[idx] * (tmp[idx] + bias[e]) + xn[idx]);
}

// td = exp(-exp(clip(time_decay + w, -9.72, 2.27)))  (in place on w)
__global__ void td_kernel(float* __restrict__ wbuf, const float* __restrict__ tdec)
{
    const size_t idx = (size_t)blockIdx.x * blockDim.x + threadIdx.x;
    if (idx >= (size_t)T_LEN * H_DIM) return;
    const int e = (int)(idx & (H_DIM - 1));
    float v = tdec[e] + wbuf[idx];
    v = fminf(fmaxf(v, -9.72f), 2.27f);
    wbuf[idx] = expf(-expf(v));
}

// ---------------------------------------------------------------------------
// Sequential WKV scan. One block per head (32 blocks, 64 threads). Thread j
// owns column j of the 64x64 state S -> 64 registers. The per-step k/r/td
// broadcast rows are staged into LDS with the CDNA5 async global->LDS engine
// (ASYNCcnt-tracked), drained with s_wait_asynccnt before each step's math.
// ---------------------------------------------------------------------------
__global__ __launch_bounds__(64) void scan_kernel(
    const float* __restrict__ kbuf, const float* __restrict__ vbuf,
    const float* __restrict__ rbuf, const float* __restrict__ td,
    const float* __restrict__ tf,   const float* __restrict__ state2_in,
    float* __restrict__ wkv, float* __restrict__ state2_out)
{
    const int h = blockIdx.x;
    const int j = threadIdx.x;
    __shared__ float ks[HSZ], rs[HSZ], ds[HSZ], fs[HSZ];

    float S[HSZ];
    const size_t hb = (size_t)h * HSZ * HSZ;
#pragma unroll
    for (int i = 0; i < HSZ; ++i) S[i] = state2_in[hb + (size_t)i * HSZ + j];
    fs[j] = tf[h * HSZ + j];

    // LDS byte offsets for this lane's slots (flat shared addr low 32 bits
    // are the LDS offset in the LDS aperture).
    const unsigned ks_off = (unsigned)(uintptr_t)&ks[j];
    const unsigned rs_off = (unsigned)(uintptr_t)&rs[j];
    const unsigned ds_off = (unsigned)(uintptr_t)&ds[j];

    __syncthreads();

    for (int t = 0; t < T_LEN; ++t) {
        const size_t rbase = (size_t)t * H_DIM + h * HSZ;
        {
            const unsigned long long ga_k = (unsigned long long)(uintptr_t)(kbuf + rbase + j);
            const unsigned long long ga_r = (unsigned long long)(uintptr_t)(rbuf + rbase + j);
            const unsigned long long ga_d = (unsigned long long)(uintptr_t)(td   + rbase + j);
            asm volatile("global_load_async_to_lds_b32 %0, %1, off"
                         :: "v"(ks_off), "v"(ga_k) : "memory");
            asm volatile("global_load_async_to_lds_b32 %0, %1, off"
                         :: "v"(rs_off), "v"(ga_r) : "memory");
            asm volatile("global_load_async_to_lds_b32 %0, %1, off"
                         :: "v"(ds_off), "v"(ga_d) : "memory");
            asm volatile("s_wait_asynccnt 0x0" ::: "memory");
        }
        __syncthreads();
        const float vj = vbuf[rbase + j];
        float acc = 0.f;
#pragma unroll
        for (int i = 0; i < HSZ; ++i) {
            const float kv = ks[i] * vj;
            acc += rs[i] * (fs[i] * kv + S[i]);
            S[i] = kv + ds[i] * S[i];
        }
        wkv[rbase + j] = acc;
        __syncthreads();
    }
#pragma unroll
    for (int i = 0; i < HSZ; ++i) state2_out[hb + (size_t)i * HSZ + j] = S[i];
}

// Group-norm over HS=64 per (t,h), scale/shift with lnx, multiply by gate g.
// One wave32 per (t,h); 8 pairs per 256-thread block.
__global__ __launch_bounds__(256) void normg_kernel(
    const float* __restrict__ wkv, const float* __restrict__ lnxw,
    const float* __restrict__ lnxb, const float* __restrict__ g,
    __bf16* __restrict__ outbf)
{
    const int lane = threadIdx.x & 31;
    const int p    = threadIdx.x >> 5;
    const int idx  = blockIdx.x * 8 + p;
    const int t = idx / NHEAD, h = idx % NHEAD;
    const size_t base = (size_t)t * H_DIM + h * HSZ;

    const float a = wkv[base + lane];
    const float c = wkv[base + lane + 32];
    float s = a + c, q = a * a + c * c;
#pragma unroll
    for (int m = 16; m > 0; m >>= 1) {
        s += __shfl_xor(s, m, 32);
        q += __shfl_xor(q, m, 32);
    }
    const float mu   = s / HSZ;
    const float rstd = rsqrtf(q / HSZ - mu * mu + EPSV);
    const int e0 = h * HSZ + lane, e1 = e0 + 32;
    const float n0 = (a - mu) * rstd * lnxw[e0] + lnxb[e0];
    const float n1 = (c - mu) * rstd * lnxw[e1] + lnxb[e1];
    outbf[base + lane]      = (__bf16)(n0 * g[base + lane]);
    outbf[base + lane + 32] = (__bf16)(n1 * g[base + lane + 32]);
}

// fp32 -> bf16 cast (same layout)
__global__ void cast_bf16(const float* __restrict__ in, __bf16* __restrict__ outp, int n)
{
    const int i = blockIdx.x * 256 + threadIdx.x;
    if (i < n) outp[i] = (__bf16)in[i];
}

// fp32 (R x C) -> bf16 (C x R)  (turn K x N weights into N x K for WMMA-B)
__global__ void transpose_cast(const float* __restrict__ in, __bf16* __restrict__ outp,
                               int R, int C)
{
    const size_t idx = (size_t)blockIdx.x * 256 + threadIdx.x;
    if (idx >= (size_t)R * C) return;
    const size_t r = idx / C, c = idx % C;
    outp[c * (size_t)R + r] = (__bf16)in[idx];
}

// ---------------------------------------------------------------------------
extern "C" void kernel_launch(void* const* d_in, const int* in_sizes, int n_in,
                              void* d_out, int out_size, void* d_ws, size_t ws_size,
                              hipStream_t stream)
{
    (void)in_sizes; (void)n_in; (void)out_size; (void)ws_size;

    const float* x        = (const float*)d_in[0];
    const float* state1   = (const float*)d_in[1];
    const float* state2   = (const float*)d_in[2];
    const float* ln1_w    = (const float*)d_in[3];
    const float* ln1_b    = (const float*)d_in[4];
    const float* tmx      = (const float*)d_in[5];
    const float* maa_w1   = (const float*)d_in[6];
    const float* maa_w2   = (const float*)d_in[7];
    const float* maa_bias = (const float*)d_in[8];
    const float* dec_w1   = (const float*)d_in[9];
    const float* dec_w2   = (const float*)d_in[10];
    const float* tdec     = (const float*)d_in[11];
    const float* tfirst   = (const float*)d_in[12];
    const float* Wr       = (const float*)d_in[13];
    const float* Wk       = (const float*)d_in[14];
    const float* Wv       = (const float*)d_in[15];
    const float* Wg       = (const float*)d_in[16];
    const float* Wo       = (const float*)d_in[17];
    const float* lnx_w    = (const float*)d_in[18];
    const float* lnx_b    = (const float*)d_in[19];

    float* out        = (float*)d_out;
    float* state1_out = out + (size_t)T_LEN * H_DIM;
    float* state2_out = state1_out + H_DIM;

    char* ws = (char*)d_ws;
    size_t off = 0;
    auto alloc = [&](size_t bytes) -> void* {
        off = (off + 255) & ~(size_t)255;
        void* p = ws + off;
        off += bytes;
        return p;
    };

    const size_t TH = (size_t)T_LEN * H_DIM;   // also H*H
    float*  xn     = (float*) alloc(TH * 4);
    float*  sx     = (float*) alloc(TH * 4);
    __bf16* xxx_bf = (__bf16*)alloc(TH * 2);
    __bf16* m_bf   = (__bf16*)alloc((size_t)T_LEN * 160 * 2);
    float*  tmp    = (float*) alloc(TH * 4);
    __bf16* mix_bf = (__bf16*)alloc(TH * 2);
    float*  kbuf   = (float*) alloc(TH * 4);
    float*  vbuf   = (float*) alloc(TH * 4);
    float*  rbuf   = (float*) alloc(TH * 4);
    float*  gbuf   = (float*) alloc(TH * 4);
    float*  wbuf   = (float*) alloc(TH * 4);
    float*  wkv    = (float*) alloc(TH * 4);
    __bf16* hdec   = (__bf16*)alloc((size_t)T_LEN * DDECC * 2);
    __bf16* ng_bf  = (__bf16*)alloc(TH * 2);
    __bf16* w1t    = (__bf16*)alloc((size_t)160 * H_DIM * 2);
    __bf16* w2t    = (__bf16*)alloc((size_t)5 * H_DIM * DMIXC * 2);
    __bf16* d1t    = (__bf16*)alloc((size_t)DDECC * H_DIM * 2);
    __bf16* d2t    = (__bf16*)alloc((size_t)H_DIM * DDECC * 2);
    __bf16* Wr_bf  = (__bf16*)alloc(TH * 2);
    __bf16* Wk_bf  = (__bf16*)alloc(TH * 2);
    __bf16* Wv_bf  = (__bf16*)alloc(TH * 2);
    __bf16* Wg_bf  = (__bf16*)alloc(TH * 2);
    __bf16* Wo_bf  = (__bf16*)alloc(TH * 2);

    const int HH = (int)TH;
    const dim3 blk256(256), blkG(128);
    const dim3 gFull(16, 16);                 // N=2048, M=2048

    // --- weight conversions -------------------------------------------------
    transpose_cast<<<dim3((H_DIM * 160 + 255) / 256), blk256, 0, stream>>>(maa_w1, w1t, H_DIM, 160);
    for (int i = 0; i < 5; ++i)
        transpose_cast<<<dim3((DMIXC * H_DIM + 255) / 256), blk256, 0, stream>>>(
            maa_w2 + (size_t)i * DMIXC * H_DIM, w2t + (size_t)i * H_DIM * DMIXC, DMIXC, H_DIM);
    transpose_cast<<<dim3((H_DIM * DDECC + 255) / 256), blk256, 0, stream>>>(dec_w1, d1t, H_DIM, DDECC);
    transpose_cast<<<dim3((DDECC * H_DIM + 255) / 256), blk256, 0, stream>>>(dec_w2, d2t, DDECC, H_DIM);
    cast_bf16<<<dim3((HH + 255) / 256), blk256, 0, stream>>>(Wr, Wr_bf, HH);
    cast_bf16<<<dim3((HH + 255) / 256), blk256, 0, stream>>>(Wk, Wk_bf, HH);
    cast_bf16<<<dim3((HH + 255) / 256), blk256, 0, stream>>>(Wv, Wv_bf, HH);
    cast_bf16<<<dim3((HH + 255) / 256), blk256, 0, stream>>>(Wg, Wg_bf, HH);
    cast_bf16<<<dim3((HH + 255) / 256), blk256, 0, stream>>>(Wo, Wo_bf, HH);

    // --- LN + token shift ---------------------------------------------------
    ln_shift_kernel<<<dim3(T_LEN), blk256, 0, stream>>>(
        x, state1, ln1_w, ln1_b, tmx, xn, sx, xxx_bf, state1_out);

    // --- m = tanh(xxx @ maa_w1)  (2048 x 160, K=2048) ----------------------
    gemm_wmma<1, false, true><<<dim3(2, 16), blkG, 0, stream>>>(
        xxx_bf, H_DIM, w1t, H_DIM, m_bf, 160, nullptr, T_LEN, 160, H_DIM);

    // --- five mix heads -----------------------------------------------------
    for (int i = 0; i < 5; ++i) {
        // tmp = m[:, i] @ maa_w2[i]   (2048 x 2048, K=32)
        gemm_wmma<0, false, false><<<gFull, blkG, 0, stream>>>(
            m_bf + i * DMIXC, 160, w2t + (size_t)i * H_DIM * DMIXC, DMIXC,
            tmp, H_DIM, nullptr, T_LEN, H_DIM, DMIXC);
        mixed_kernel<<<dim3((unsigned)((TH + 255) / 256)), blk256, 0, stream>>>(
            tmp, maa_bias + (size_t)i * H_DIM, sx, xn, mix_bf);

        switch (i) {
        case 0:  // mw -> decay MLP -> td
            gemm_wmma<1, false, true><<<dim3(1, 16), blkG, 0, stream>>>(
                mix_bf, H_DIM, d1t, H_DIM, hdec, DDECC, nullptr, T_LEN, DDECC, H_DIM);
            gemm_wmma<0, false, false><<<gFull, blkG, 0, stream>>>(
                hdec, DDECC, d2t, DDECC, wbuf, H_DIM, nullptr, T_LEN, H_DIM, DDECC);
            td_kernel<<<dim3((unsigned)((TH + 255) / 256)), blk256, 0, stream>>>(wbuf, tdec);
            break;
        case 1:  // mk -> k
            gemm_wmma<0, false, false><<<gFull, blkG, 0, stream>>>(
                mix_bf, H_DIM, Wk_bf, H_DIM, kbuf, H_DIM, nullptr, T_LEN, H_DIM, H_DIM);
            break;
        case 2:  // mv -> v
            gemm_wmma<0, false, false><<<gFull, blkG, 0, stream>>>(
                mix_bf, H_DIM, Wv_bf, H_DIM, vbuf, H_DIM, nullptr, T_LEN, H_DIM, H_DIM);
            break;
        case 3:  // mr -> r
            gemm_wmma<0, false, false><<<gFull, blkG, 0, stream>>>(
                mix_bf, H_DIM, Wr_bf, H_DIM, rbuf, H_DIM, nullptr, T_LEN, H_DIM, H_DIM);
            break;
        case 4:  // mg -> g = silu(.)
            gemm_wmma<2, false, false><<<gFull, blkG, 0, stream>>>(
                mix_bf, H_DIM, Wg_bf, H_DIM, gbuf, H_DIM, nullptr, T_LEN, H_DIM, H_DIM);
            break;
        }
    }

    // --- sequential WKV scan -----------------------------------------------
    scan_kernel<<<dim3(NHEAD), dim3(HSZ), 0, stream>>>(
        kbuf, vbuf, rbuf, wbuf, tfirst, state2, wkv, state2_out);

    // --- group-norm * gate -> bf16 ------------------------------------------
    normg_kernel<<<dim3(T_LEN * NHEAD / 8), blk256, 0, stream>>>(
        wkv, lnx_w, lnx_b, gbuf, ng_bf);

    // --- out = (normed*g) @ Wo.T + x ----------------------------------------
    gemm_wmma<0, true, false><<<gFull, blkG, 0, stream>>>(
        ng_bf, H_DIM, Wo_bf, H_DIM, out, H_DIM, x, T_LEN, H_DIM, H_DIM);
}
